// MultiHeadAttention_88132728914247
// MI455X (gfx1250) — matrix operand order repaired and checked by reading, not compiled, not run on hardware
//
#include <hip/hip_runtime.h>

// ---------------------------------------------------------------------------
// Multi-head attention + residual + LayerNorm, CDNA5 (gfx1250) WMMA bf16 path.
//   B=2, S=2048, D_MODEL=1024, H=16, d_head=64
// Pipeline: 3x QKV GEMM (WMMA, LDS-staged) -> flash attention (WMMA, async
// double-buffered LDS tiles) -> out-proj GEMM (+bias+residual) -> LayerNorm.
// ---------------------------------------------------------------------------

#define B_   2
#define S_   2048
#define DM_  1024
#define H_   16
#define DH_  64
#define NEGF (-1e9f)
#define LN_EPS 1e-5f

typedef __attribute__((ext_vector_type(16))) __bf16 bf16x16;
typedef __attribute__((ext_vector_type(8)))  float  f32x8;
typedef __attribute__((address_space(3)))    char   lds_char;

// ---------------------------------------------------------------------------
// gfx1250 async global->LDS copy (ASYNCcnt-tracked).  VDST carries the LDS
// byte offset, VADDR the 64-bit global address (GV mode).
// ---------------------------------------------------------------------------
__device__ __forceinline__ void async_load_b128(void* lds, const void* gaddr) {
  unsigned loff = (unsigned)(size_t)(lds_char*)lds;   // generic -> LDS offset
  asm volatile("global_load_async_to_lds_b128 %0, %1, off"
               :: "v"(loff), "v"(gaddr) : "memory");
}
__device__ __forceinline__ void wait_async0() {
  asm volatile("s_wait_asynccnt 0x0" ::: "memory");
}

// ---------------------------------------------------------------------------
// Fragment loaders for CDNA5 WMMA (wave32); A and B share a contiguous K order, not the ISA table order.
// A (16x32 bf16): lane l holds row M=l&15; element e -> K = e + 16*(l>>4)
// B (32x16 bf16): lane l holds col N=l&15; element e -> K = e + 16*(l>>4)
// C/D (16x16 f32): reg r -> M = r + 8*(l>>4), N = l&15
// ---------------------------------------------------------------------------
__device__ __forceinline__ bf16x16 load_a_frag(const __bf16* A, int ld) {
  const int lane = threadIdx.x & 31;
  const int row  = lane & 15;
  const int half = lane >> 4;
  const __bf16* base = A + row * ld;
  union { bf16x16 v; unsigned int u[8]; } r;
#pragma unroll
  for (int g = 0; g < 8; ++g) {
    int kbase = 2 * g + 16 * half;
    r.u[g] = *reinterpret_cast<const unsigned int*>(base + kbase);
  }
  return r.v;
}

// K-contiguous source: Bt laid out [n][k], column stride ld elements.
__device__ __forceinline__ bf16x16 load_b_frag(const __bf16* Bt, int ld) {
  const int lane = threadIdx.x & 31;
  const int col  = lane & 15;
  const int half = lane >> 4;
  const unsigned int* src =
      reinterpret_cast<const unsigned int*>(Bt + col * ld + 16 * half);
  union { bf16x16 v; unsigned int u[8]; } r;
#pragma unroll
  for (int g = 0; g < 8; ++g) r.u[g] = src[g];
  return r.v;
}

// Reductions across the 16 lanes that share a C/D matrix row (wave32).
__device__ __forceinline__ float rowmax16(float v) {
#pragma unroll
  for (int m = 1; m < 16; m <<= 1) v = fmaxf(v, __shfl_xor(v, m, 32));
  return v;
}
__device__ __forceinline__ float rowsum16(float v) {
#pragma unroll
  for (int m = 1; m < 16; m <<= 1) v += __shfl_xor(v, m, 32);
  return v;
}

// ---------------------------------------------------------------------------
// Generic GEMM: C[M,N] = A(fp32,[M,K]) * W(fp32,[K,N]) + bias; A/W converted
// to bf16 while staging through LDS.  Block tile 128x64, 8 waves, K-step 32.
//   mode 0: fp32 out[row*N+col] (+ residual)
//   mode 1: bf16 scatter to [B,H,S,DH]   (Q,K path)
//   mode 2: bf16 scatter to [B,H,DH,S]   (V path, pre-transposed for attn)
// ---------------------------------------------------------------------------
__global__ __launch_bounds__(256)
void gemm_bf16_wmma(const float* __restrict__ A, const float* __restrict__ W,
                    const float* __restrict__ bias,
                    const float* __restrict__ residual,
                    float* __restrict__ outF, __bf16* __restrict__ outHead,
                    int M, int N, int K, int mode) {
  __shared__ __align__(16) __bf16 ldsA[128 * 32];   // [m][k]
  __shared__ __align__(16) __bf16 ldsBt[64 * 32];   // [n][k] (K-contiguous)

  const int tid  = threadIdx.x;
  const int wave = tid >> 5;
  const int lane = tid & 31;
  const int half = lane >> 4;
  const int rowBase = blockIdx.y * 128;
  const int colBase = blockIdx.x * 64;

  f32x8 acc[4] = {};

  for (int k0 = 0; k0 < K; k0 += 32) {
    if (k0 + 32 < K) {   // cover L2->WGP latency for the next K tile
      __builtin_prefetch(&A[(size_t)(rowBase + (tid & 127)) * K + k0 + 32], 0, 3);
      __builtin_prefetch(&W[(size_t)(k0 + 32 + (tid & 31)) * N + colBase], 0, 3);
    }
#pragma unroll 4
    for (int idx = tid; idx < 128 * 32; idx += 256) {
      int m = idx >> 5, kk = idx & 31;
      ldsA[idx] = (__bf16)A[(size_t)(rowBase + m) * K + k0 + kk];
    }
#pragma unroll 2
    for (int idx = tid; idx < 32 * 64; idx += 256) {
      int kk = idx >> 6, n = idx & 63;
      ldsBt[n * 32 + kk] = (__bf16)W[(size_t)(k0 + kk) * N + colBase + n];
    }
    __syncthreads();

    bf16x16 a = load_a_frag(ldsA + wave * 16 * 32, 32);
#pragma unroll
    for (int n = 0; n < 4; ++n) {
      bf16x16 b = load_b_frag(ldsBt + n * 16 * 32, 32);
      acc[n] = __builtin_amdgcn_wmma_f32_16x16x32_bf16(
          false, a, false, b, (short)0, acc[n], false, false);
    }
    __syncthreads();
  }

#pragma unroll
  for (int n = 0; n < 4; ++n) {
#pragma unroll
    for (int r = 0; r < 8; ++r) {
      int row = rowBase + wave * 16 + r + 8 * half;
      int col = colBase + n * 16 + (lane & 15);
      float v = acc[n][r] + bias[col];
      if (mode == 0) {
        if (residual) v += residual[(size_t)row * N + col];
        outF[(size_t)row * N + col] = v;
      } else {
        int b = row >> 11, s = row & (S_ - 1);        // M = B*S
        int h = col >> 6,  d = col & (DH_ - 1);       // N = H*DH
        size_t hb = (size_t)(b * H_ + h);
        if (mode == 1)
          outHead[(hb * S_ + s) * DH_ + d] = (__bf16)v;        // [B,H,S,DH]
        else
          outHead[(hb * DH_ + d) * S_ + s] = (__bf16)v;        // [B,H,DH,S]
      }
    }
  }
}

// ---------------------------------------------------------------------------
// Flash attention: one block = (b,h, 64 query rows); 4 waves, each owns a
// 16-row query tile.  32-key tiles are streamed into double-buffered LDS with
// gfx1250 async global->LDS B128 copies issued one tile ahead; a single
// barrier per tile separates arrival from the next issue.
// ---------------------------------------------------------------------------

// K tile: [32 keys][64 d] bf16 (4KB, 256 x 16B chunks), straight copy.
// V tile: [64 d][32 keys] bf16 (4KB) -- straight copy because V is stored
// pre-transposed per head in global ([B,H,DH,S]).
__device__ __forceinline__ void issue_tile_copies(
    const char* kbase, const char* vtbase, int key0,
    __bf16* ldsKbuf, __bf16* ldsVbuf, int tid) {
#pragma unroll
  for (int j = 0; j < 2; ++j) {
    int c = tid + j * 128;                          // chunk id 0..255
    const char* gk = kbase + (size_t)(key0 + (c >> 3)) * (DH_ * 2) + (c & 7) * 16;
    async_load_b128((char*)ldsKbuf + c * 16, gk);
    const char* gv = vtbase + (size_t)(c >> 2) * (S_ * 2) + (size_t)key0 * 2 + (c & 3) * 16;
    async_load_b128((char*)ldsVbuf + c * 16, gv);
  }
}

__global__ __launch_bounds__(128)
void flash_attn_wmma(const __bf16* __restrict__ qg, const __bf16* __restrict__ kg,
                     const __bf16* __restrict__ vtg,
                     const unsigned char* __restrict__ mask,
                     float* __restrict__ ctx) {
  __shared__ __align__(16) __bf16 ldsK[2][32 * 64];   // [key][d]
  __shared__ __align__(16) __bf16 ldsVt[2][64 * 32];  // [d][key]
  __shared__ __align__(16) __bf16 ldsP[4][16 * 32];   // per-wave P tile

  const int tid  = threadIdx.x;
  const int wave = tid >> 5;
  const int lane = tid & 31;
  const int half = lane >> 4;
  const int bh = blockIdx.y;
  const int b = bh / H_, h = bh % H_;
  const int q0 = blockIdx.x * 64 + wave * 16;

  const size_t headBase = ((size_t)(b * H_ + h)) * S_ * DH_;
  const char* kbase  = (const char*)(kg  + headBase);   // [S][DH] bf16
  const char* vtbase = (const char*)(vtg + headBase);   // [DH][S] bf16

  const __bf16* qbase = qg + headBase + (size_t)q0 * DH_;
  const bf16x16 qf0 = load_a_frag(qbase, DH_);        // K = 0..31 of d_head
  const bf16x16 qf1 = load_a_frag(qbase + 32, DH_);   // K = 32..63

  f32x8 oacc[4] = {};
  float mst[8], lst[8];
#pragma unroll
  for (int r = 0; r < 8; ++r) { mst[r] = -3.0e38f; lst[r] = 0.f; }

  const unsigned char* mbase = mask + (size_t)b * S_ * S_;

  // Prologue: stage tile 0.
  issue_tile_copies(kbase, vtbase, 0, ldsK[0], ldsVt[0], tid);

  const int NT = S_ / 32;
  for (int it = 0; it < NT; ++it) {
    const int cur  = it & 1;
    const int key0 = it * 32;

    wait_async0();        // this wave's copies of tile `it` have landed
    __syncthreads();      // whole tile visible; prev tile's readers are done

    if (it + 1 < NT)      // stream next tile into the other buffer
      issue_tile_copies(kbase, vtbase, key0 + 32, ldsK[cur ^ 1], ldsVt[cur ^ 1], tid);

    // Scores: 16x32 tile = two 16x16 WMMA outputs, K=64 in two 32-chunks.
    f32x8 s0 = {}, s1 = {};
    {
      bf16x16 b00 = load_b_frag(ldsK[cur] + 0 * 16 * 64 + 0, 64);
      bf16x16 b01 = load_b_frag(ldsK[cur] + 0 * 16 * 64 + 32, 64);
      s0 = __builtin_amdgcn_wmma_f32_16x16x32_bf16(false, qf0, false, b00, (short)0, s0, false, false);
      s0 = __builtin_amdgcn_wmma_f32_16x16x32_bf16(false, qf1, false, b01, (short)0, s0, false, false);
      bf16x16 b10 = load_b_frag(ldsK[cur] + 1 * 16 * 64 + 0, 64);
      bf16x16 b11 = load_b_frag(ldsK[cur] + 1 * 16 * 64 + 32, 64);
      s1 = __builtin_amdgcn_wmma_f32_16x16x32_bf16(false, qf0, false, b10, (short)0, s1, false, false);
      s1 = __builtin_amdgcn_wmma_f32_16x16x32_bf16(false, qf1, false, b11, (short)0, s1, false, false);
    }

    // Scale + mask + online softmax (D-layout row r -> M = r + 8*half).
#pragma unroll
    for (int r = 0; r < 8; ++r) {
      int qrow = q0 + r + 8 * half;
      const unsigned char* mr = mbase + (size_t)qrow * S_ + key0 + (lane & 15);
      float v0 = mr[0]  ? NEGF : s0[r] * 0.125f;   // 1/sqrt(64)
      float v1 = mr[16] ? NEGF : s1[r] * 0.125f;
      float mn = fmaxf(mst[r], rowmax16(fmaxf(v0, v1)));
      float alpha = __expf(mst[r] - mn);
      float p0 = __expf(v0 - mn);
      float p1 = __expf(v1 - mn);
      lst[r] = lst[r] * alpha + rowsum16(p0 + p1);
      mst[r] = mn;
#pragma unroll
      for (int n = 0; n < 4; ++n) oacc[n][r] *= alpha;
      // D-layout -> row-major P tile in this wave's LDS slice.
      ldsP[wave][(r + 8 * half) * 32 + (lane & 15)]      = (__bf16)p0;
      ldsP[wave][(r + 8 * half) * 32 + (lane & 15) + 16] = (__bf16)p1;
    }
    __builtin_amdgcn_wave_barrier();   // keep same-wave LDS ops ordered

    // Context: oacc += P(16x32) x V(32x64)
    bf16x16 pf = load_a_frag(&ldsP[wave][0], 32);
#pragma unroll
    for (int n = 0; n < 4; ++n) {
      bf16x16 bv = load_b_frag(ldsVt[cur] + n * 16 * 32, 32);
      oacc[n] = __builtin_amdgcn_wmma_f32_16x16x32_bf16(
          false, pf, false, bv, (short)0, oacc[n], false, false);
    }
    // No trailing barrier: next iteration's wait+barrier gates buffer reuse.
  }

  // ctx[b, s, h*64 + d] = o / l   (fp32 workspace for the output projection)
#pragma unroll
  for (int n = 0; n < 4; ++n) {
#pragma unroll
    for (int r = 0; r < 8; ++r) {
      int row = q0 + r + 8 * half;
      int col = h * DH_ + n * 16 + (lane & 15);
      ctx[((size_t)b * S_ + row) * DM_ + col] = oacc[n][r] / lst[r];
    }
  }
}

// ---------------------------------------------------------------------------
// LayerNorm over the last dim (1024), one block per row.
// ---------------------------------------------------------------------------
__global__ __launch_bounds__(256)
void layernorm_rows(const float* __restrict__ x, const float* __restrict__ gamma,
                    const float* __restrict__ beta, float* __restrict__ out) {
  __shared__ float sh[256];
  const int row = blockIdx.x;
  const int tid = threadIdx.x;
  const float* xr = x + (size_t)row * DM_;

  float v[4];
  float s = 0.f, s2 = 0.f;
#pragma unroll
  for (int j = 0; j < 4; ++j) {
    v[j] = xr[tid + j * 256];
    s += v[j];
    s2 += v[j] * v[j];
  }
  sh[tid] = s; __syncthreads();
  for (int off = 128; off > 0; off >>= 1) {
    if (tid < off) sh[tid] += sh[tid + off];
    __syncthreads();
  }
  float total = sh[0]; __syncthreads();
  sh[tid] = s2; __syncthreads();
  for (int off = 128; off > 0; off >>= 1) {
    if (tid < off) sh[tid] += sh[tid + off];
    __syncthreads();
  }
  float total2 = sh[0];

  float mu   = total * (1.0f / DM_);
  float var  = total2 * (1.0f / DM_) - mu * mu;
  float rstd = rsqrtf(var + LN_EPS);
#pragma unroll
  for (int j = 0; j < 4; ++j) {
    int c = tid + j * 256;
    out[(size_t)row * DM_ + c] = (v[j] - mu) * rstd * gamma[c] + beta[c];
  }
}

// ---------------------------------------------------------------------------
// Launch wiring.  Inputs (setup_inputs order):
//  0 Q f32[B,S,DM], 1 attn_mask bool[B,S,S], 2 batch_size i32,
//  3 Wq, 4 bq, 5 Wk, 6 bk, 7 Wv, 8 bv, 9 Wo, 10 bo, 11 gamma, 12 beta
// ---------------------------------------------------------------------------
extern "C" void kernel_launch(void* const* d_in, const int* in_sizes, int n_in,
                              void* d_out, int out_size, void* d_ws, size_t ws_size,
                              hipStream_t stream) {
  (void)in_sizes; (void)n_in; (void)out_size; (void)ws_size;

  const float* Q    = (const float*)d_in[0];
  const unsigned char* mask = (const unsigned char*)d_in[1];  // jax bool = 1 byte
  const float* Wq   = (const float*)d_in[3];
  const float* bq   = (const float*)d_in[4];
  const float* Wk   = (const float*)d_in[5];
  const float* bk   = (const float*)d_in[6];
  const float* Wv   = (const float*)d_in[7];
  const float* bv   = (const float*)d_in[8];
  const float* Wo   = (const float*)d_in[9];
  const float* bo   = (const float*)d_in[10];
  const float* gamma= (const float*)d_in[11];
  const float* beta = (const float*)d_in[12];
  float* out = (float*)d_out;

  // Workspace layout (56 MB):
  char* ws = (char*)d_ws;
  __bf16* qb  = (__bf16*)(ws);                          //  8 MB  [B,H,S,DH]
  __bf16* kb  = (__bf16*)(ws + ((size_t)8  << 20));     //  8 MB  [B,H,S,DH]
  __bf16* vtb = (__bf16*)(ws + ((size_t)16 << 20));     //  8 MB  [B,H,DH,S]
  float*  ctx = (float*) (ws + ((size_t)24 << 20));     // 16 MB  [B,S,DM]
  float*  pre = (float*) (ws + ((size_t)40 << 20));     // 16 MB  [B,S,DM]

  const int M = B_ * S_;          // 4096
  const int N = DM_;              // 1024
  const int K = DM_;              // 1024
  dim3 gemmGrid(N / 64, M / 128); // (16, 32)

  // QKV projections -> bf16 per-head layouts (V transposed for async staging)
  gemm_bf16_wmma<<<gemmGrid, 256, 0, stream>>>(Q, Wq, bq, nullptr, nullptr, qb,  M, N, K, 1);
  gemm_bf16_wmma<<<gemmGrid, 256, 0, stream>>>(Q, Wk, bk, nullptr, nullptr, kb,  M, N, K, 1);
  gemm_bf16_wmma<<<gemmGrid, 256, 0, stream>>>(Q, Wv, bv, nullptr, nullptr, vtb, M, N, K, 2);

  // Flash attention -> fp32 context [B,S,H*DH]
  flash_attn_wmma<<<dim3(S_ / 64, B_ * H_), 128, 0, stream>>>(qb, kb, vtb, mask, ctx);

  // Output projection + bias + residual -> pre-LN activations
  gemm_bf16_wmma<<<gemmGrid, 256, 0, stream>>>(ctx, Wo, bo, Q, pre, nullptr, M, N, K, 0);

  // LayerNorm -> final output
  layernorm_rows<<<M, 256, 0, stream>>>(pre, gamma, beta, out);
}
